// FieldAttentionalRNNDecoder_26869315404117
// MI455X (gfx1250) — matrix operand order repaired and checked by reading, not compiled, not run on hardware
//
#include <hip/hip_runtime.h>
#include <hip/hip_bf16.h>
#include <math.h>

// ---------------------------------------------------------------------------
// Problem constants (from reference)
// ---------------------------------------------------------------------------
#define BB 64
#define SS 100
#define TT 50
#define HH 512
#define EE 512
#define FF 64
#define VV 32000
#define TP1 (TT + 1)   // 51 decode steps

typedef __attribute__((ext_vector_type(16))) __bf16 bf16x16;
typedef __attribute__((ext_vector_type(8)))  float  floatx8;

__device__ __forceinline__ unsigned short f2bf(float f) {
    // round-to-nearest-even f32 -> bf16
    unsigned u = __float_as_uint(f);
    unsigned r = ((u >> 16) & 1u) + 0x7FFFu;
    return (unsigned short)((u + r) >> 16);
}

__device__ __forceinline__ float sigf(float x) { return 1.0f / (1.0f + __expf(-x)); }

// ---------------------------------------------------------------------------
// Pack a row-major f32 weight W[K][N] into WMMA B-fragment panels (bf16).
// Tile = 32(K) x 16(N), panel index = nt*nKt + kt, 512 bf16 per panel.
// Element e of lane l holds W[kt*32 + (e<8 ? e + 8*(l>>4) : e + 8 + 8*(l>>4))][nt*16 + (l&15)]
// (B 32x16 layout: same k order as the A fragment).
// ---------------------------------------------------------------------------
__global__ void pack_b_kernel(const float* __restrict__ W, unsigned short* __restrict__ out,
                              int K, int N) {
    long idx = (long)blockIdx.x * blockDim.x + threadIdx.x;
    long total = (long)K * N;
    if (idx >= total) return;
    int  nKt    = K >> 5;
    long tile   = idx >> 9;
    int  within = (int)(idx & 511);
    int  lane   = within >> 4;
    int  e      = within & 15;
    long kt  = tile % nKt;
    long ntv = tile / nKt;
    int n = (int)(ntv << 4) + (lane & 15);
    int k = (int)(kt  << 5) + ((e < 8) ? (e + ((lane >> 4) << 3)) : (e + 8 + ((lane >> 4) << 3)));
    out[idx] = f2bf(W[(long)k * N + n]);
}

// ---------------------------------------------------------------------------
// Pack an f32 activation matrix (optionally the concat of two sources along K,
// with per-source row stride; lda0 == 0 broadcasts one row) into WMMA
// A-fragment panels (bf16). Tile = 16(M) x 32(K), panel index = mt*nKt + kt.
// A 16x32 layout (§7.12.2): lane m = l&15, half = l>>4,
//   element e<8  -> k = e + 8*half
//   element e>=8 -> k = e + 8 + 8*half
// ---------------------------------------------------------------------------
__global__ void pack_a_kernel(const float* __restrict__ A0, int lda0,
                              const float* __restrict__ A1, int lda1, int K0,
                              unsigned short* __restrict__ out, int M, int K) {
    long idx = (long)blockIdx.x * blockDim.x + threadIdx.x;
    long total = (long)M * K;
    if (idx >= total) return;
    int  nKt    = K >> 5;
    long tile   = idx >> 9;
    int  within = (int)(idx & 511);
    int  lane   = within >> 4;
    int  e      = within & 15;
    long kt = tile % nKt;
    long mt = tile / nKt;
    int  m    = (int)(mt << 4) + (lane & 15);
    int  half = lane >> 4;
    int  k    = (int)(kt << 5) + ((e < 8) ? (e + (half << 3)) : (e + 8 + (half << 3)));
    float v = (k < K0) ? A0[(long)m * lda0 + k]
                       : A1[(long)m * lda1 + (k - K0)];
    out[idx] = f2bf(v);
}

// ---------------------------------------------------------------------------
// Fully packed bf16 WMMA GEMM: C = act(Ap @ Bp + bias).
// One wave computes a 16(M) x 64(N) strip: 4 accumulators, 4 v_wmma per
// k-step. One-deep software pipeline: fragments for k-step kt+1 are fetched
// (global_load_b128 pairs) BEFORE the 4 WMMAs of k-step kt execute, so the
// waits become loadcnt<=outstanding instead of loadcnt==0 and the next fetch
// overlaps the matrix ops.
//   act: 0 = none, 1 = tanh
//   remapT: if >0, logical row m = t*64+b is stored at row b*remapT + t
// ---------------------------------------------------------------------------
__global__ void wmma_gemm_pp(const unsigned short* __restrict__ Ap,
                             const unsigned short* __restrict__ Bp,
                             const float* __restrict__ bias,
                             float* __restrict__ C, int ldc,
                             int M, int N, int K, int act, int remapT) {
    const int lane    = threadIdx.x & 31;
    const int nGroups = N >> 6;                      // groups of 4 n-tiles
    const int tiles   = (M >> 4) * nGroups;
    int wt = blockIdx.x * 8 + (threadIdx.x >> 5);
    if (wt >= tiles) return;                         // wave-uniform: EXEC stays all-1s
    const int  mt      = wt / nGroups;
    const int  nt0     = (wt - mt * nGroups) << 2;
    const int  nKt     = K >> 5;                     // >= 2 for all shapes here
    const long laneOff = (long)(lane << 4);
    const long bStride = (long)nKt << 9;             // elements between n-tile panels

    const unsigned short* ap = Ap + (((long)mt * nKt) << 9) + laneOff;
    const unsigned short* bp = Bp + ((long)nt0 * bStride) + laneOff;

    floatx8 acc[4];
#pragma unroll
    for (int j = 0; j < 4; ++j)
        acc[j] = (floatx8){0.f, 0.f, 0.f, 0.f, 0.f, 0.f, 0.f, 0.f};

    // prologue: fragments for k-step 0
    bf16x16 a  = *(const bf16x16*)(ap);
    bf16x16 b0 = *(const bf16x16*)(bp);
    bf16x16 b1 = *(const bf16x16*)(bp + bStride);
    bf16x16 b2 = *(const bf16x16*)(bp + 2 * bStride);
    bf16x16 b3 = *(const bf16x16*)(bp + 3 * bStride);

    for (int kt = 0; kt < nKt - 1; ++kt) {
        const long ko = (long)(kt + 1) << 9;
        // issue next k-step's loads first ...
        bf16x16 an  = *(const bf16x16*)(ap + ko);
        bf16x16 b0n = *(const bf16x16*)(bp + ko);
        bf16x16 b1n = *(const bf16x16*)(bp + bStride + ko);
        bf16x16 b2n = *(const bf16x16*)(bp + 2 * bStride + ko);
        bf16x16 b3n = *(const bf16x16*)(bp + 3 * bStride + ko);
        // ... then consume the current (already-arrived) fragments
        acc[0] = __builtin_amdgcn_wmma_f32_16x16x32_bf16(false, a, false, b0, (short)0, acc[0], false, false);
        acc[1] = __builtin_amdgcn_wmma_f32_16x16x32_bf16(false, a, false, b1, (short)0, acc[1], false, false);
        acc[2] = __builtin_amdgcn_wmma_f32_16x16x32_bf16(false, a, false, b2, (short)0, acc[2], false, false);
        acc[3] = __builtin_amdgcn_wmma_f32_16x16x32_bf16(false, a, false, b3, (short)0, acc[3], false, false);
        a = an; b0 = b0n; b1 = b1n; b2 = b2n; b3 = b3n;
    }
    // epilogue: last k-step
    acc[0] = __builtin_amdgcn_wmma_f32_16x16x32_bf16(false, a, false, b0, (short)0, acc[0], false, false);
    acc[1] = __builtin_amdgcn_wmma_f32_16x16x32_bf16(false, a, false, b1, (short)0, acc[1], false, false);
    acc[2] = __builtin_amdgcn_wmma_f32_16x16x32_bf16(false, a, false, b2, (short)0, acc[2], false, false);
    acc[3] = __builtin_amdgcn_wmma_f32_16x16x32_bf16(false, a, false, b3, (short)0, acc[3], false, false);

    const int half = lane >> 4;
#pragma unroll
    for (int j = 0; j < 4; ++j) {
        const int   n   = ((nt0 + j) << 4) + (lane & 15);
        const float bsv = bias[n];
#pragma unroll
        for (int r = 0; r < 8; ++r) {
            int   mm = (mt << 4) + r + (half << 3);
            float v  = acc[j][r] + bsv;
            if (act) v = tanhf(v);
            long orow = remapT ? ((long)(mm & 63) * remapT + (mm >> 6)) : (long)mm;
            C[orow * (long)ldc + n] = v;
        }
    }
}

// ---------------------------------------------------------------------------
// LSTM elementwise: z[64,2048] split into i|j|f|o; update c,h in place.
// ---------------------------------------------------------------------------
__global__ void lstm_ew_kernel(const float* __restrict__ z,
                               float* __restrict__ h, float* __restrict__ c) {
    int idx = blockIdx.x * blockDim.x + threadIdx.x;
    if (idx >= BB * HH) return;
    int b = idx >> 9, n = idx & 511;
    const float* zr = z + (long)b * (4 * HH);
    float iv = zr[n], jv = zr[n + HH], fv = zr[n + 2 * HH], ov = zr[n + 3 * HH];
    float cp = c[idx];
    float cn = sigf(fv + 1.0f) * cp + sigf(iv) * tanhf(jv);
    c[idx] = cn;
    h[idx] = sigf(ov) * tanhf(cn);
}

// ---------------------------------------------------------------------------
// Dual attention for one step: block b handles batch row b.
// ---------------------------------------------------------------------------
__global__ void attention_kernel(const float* __restrict__ phi_h,
                                 const float* __restrict__ phi_f,
                                 const float* __restrict__ enc,
                                 const float* __restrict__ gamma,
                                 const float* __restrict__ alpha,
                                 float* __restrict__ context,
                                 float* __restrict__ atts, int t) {
    const int b   = blockIdx.x;
    const int tid = threadIdx.x;
    __shared__ float w[SS], fw[SS], wc[SS];

    for (int s = tid; s < SS; s += blockDim.x) {
        const float* ph = phi_h + ((long)b * SS + s) * HH;
        const float* pf = phi_f + ((long)b * SS + s) * HH;
        const float* g  = gamma + (long)b * HH;
        const float* al = alpha + (long)b * HH;
        float aw = 0.f, af = 0.f;
        for (int k = 0; k < HH; ++k) { aw += ph[k] * g[k]; af += pf[k] * al[k]; }
        w[s] = aw; fw[s] = af;
    }
    __syncthreads();

    if (tid == 0) {
        float wm = -1e30f, fm = -1e30f;
        for (int s = 0; s < SS; ++s) { wm = fmaxf(wm, w[s]); fm = fmaxf(fm, fw[s]); }
        float wsum = 0.f, fsum = 0.f;
        for (int s = 0; s < SS; ++s) {
            w[s]  = __expf(w[s]  - wm); wsum += w[s];
            fw[s] = __expf(fw[s] - fm); fsum += fw[s];
        }
        float iw = 1.f / (1e-6f + wsum), jf = 1.f / (1e-6f + fsum);
        float cs = 0.f;
        for (int s = 0; s < SS; ++s) { wc[s] = (w[s] * iw) * (fw[s] * jf); cs += wc[s]; }
        float ic = 1.f / (1e-6f + cs);
        for (int s = 0; s < SS; ++s) wc[s] *= ic;
    }
    __syncthreads();

    for (int s = tid; s < SS; s += blockDim.x)
        atts[(long)t * (SS * BB) + (long)s * BB + b] = wc[s];

    for (int k = tid; k < HH; k += blockDim.x) {
        float a = 0.f;
        const float* eb = enc + (long)b * SS * HH + k;
        for (int s = 0; s < SS; ++s) a += eb[(long)s * HH] * wc[s];
        context[(long)b * HH + k] = a;
    }
}

// ---------------------------------------------------------------------------
// Host-side orchestration
// ---------------------------------------------------------------------------
static inline void launch_gemm(hipStream_t st, const unsigned short* Ap,
                               const unsigned short* Bp, const float* bias,
                               float* C, int ldc, int M, int N, int K,
                               int act, int remapT) {
    int tiles  = (M >> 4) * (N >> 6);
    int blocks = (tiles + 7) >> 3;   // 8 waves (16x64 strips) per 256-thread block
    hipLaunchKernelGGL(wmma_gemm_pp, dim3(blocks), dim3(256), 0, st,
                       Ap, Bp, bias, C, ldc, M, N, K, act, remapT);
}

static inline void launch_pack_b(hipStream_t st, const float* W, unsigned short* out,
                                 int K, int N) {
    long total = (long)K * N;
    hipLaunchKernelGGL(pack_b_kernel, dim3((int)((total + 255) / 256)), dim3(256), 0,
                       st, W, out, K, N);
}

static inline void launch_pack_a(hipStream_t st, const float* A0, int lda0,
                                 const float* A1, int lda1, int K0,
                                 unsigned short* out, int M, int K) {
    long total = (long)M * K;
    hipLaunchKernelGGL(pack_a_kernel, dim3((int)((total + 255) / 256)), dim3(256), 0,
                       st, A0, lda0, A1, lda1, K0, out, M, K);
}

extern "C" void kernel_launch(void* const* d_in, const int* in_sizes, int n_in,
                              void* d_out, int out_size, void* d_ws, size_t ws_size,
                              hipStream_t stream) {
    const float* h0   = (const float*)d_in[0];
    const float* c0   = (const float*)d_in[1];
    const float* emb  = (const float*)d_in[2];
    const float* enc  = (const float*)d_in[3];
    const float* fld  = (const float*)d_in[4];
    const float* iemb = (const float*)d_in[5];
    const float* W1 = (const float*)d_in[6],  *b1 = (const float*)d_in[7];
    const float* W2 = (const float*)d_in[8],  *b2 = (const float*)d_in[9];
    const float* W3 = (const float*)d_in[10], *b3 = (const float*)d_in[11];
    const float* W4 = (const float*)d_in[12], *b4 = (const float*)d_in[13];
    const float* W5 = (const float*)d_in[14], *b5 = (const float*)d_in[15];
    const float* Wl = (const float*)d_in[16], *bl = (const float*)d_in[17];
    const float* Wo = (const float*)d_in[18], *bo = (const float*)d_in[19];

    // ---- workspace carve-up (all rewritten every call; deterministic) ----
    char* wp = (char*)d_ws;
    auto carve = [&](size_t bytes) -> char* {
        char* p = wp;
        wp += (bytes + 255) & ~(size_t)255;
        return p;
    };
    // packed weights (B panels, bf16)
    unsigned short* WoP = (unsigned short*)carve((size_t)HH * VV * 2);
    unsigned short* WlP = (unsigned short*)carve((size_t)(EE + HH) * 4 * HH * 2);
    unsigned short* W1P = (unsigned short*)carve((size_t)HH * HH * 2);
    unsigned short* W2P = (unsigned short*)carve((size_t)HH * HH * 2);
    unsigned short* W3P = (unsigned short*)carve((size_t)2 * HH * HH * 2);
    unsigned short* W4P = (unsigned short*)carve((size_t)FF * HH * 2);
    unsigned short* W5P = (unsigned short*)carve((size_t)HH * HH * 2);
    // packed activations (A panels, bf16)
    unsigned short* encAP = (unsigned short*)carve((size_t)BB * SS * HH * 2);
    unsigned short* fldAP = (unsigned short*)carve((size_t)BB * SS * FF * 2);
    unsigned short* zAP   = (unsigned short*)carve((size_t)BB * (EE + HH) * 2);
    unsigned short* hAP   = (unsigned short*)carve((size_t)BB * HH * 2);
    unsigned short* catAP = (unsigned short*)carve((size_t)BB * 2 * HH * 2);
    unsigned short* attAP = (unsigned short*)carve((size_t)TP1 * BB * HH * 2);
    // f32 buffers
    float* phi_h   = (float*)carve((size_t)BB * SS * HH * 4);
    float* phi_f   = (float*)carve((size_t)BB * SS * HH * 4);
    float* hbuf    = (float*)carve((size_t)BB * HH * 4);
    float* cbuf    = (float*)carve((size_t)BB * HH * 4);
    float* zbuf    = (float*)carve((size_t)BB * 4 * HH * 4);
    float* gbuf    = (float*)carve((size_t)BB * HH * 4);
    float* abuf    = (float*)carve((size_t)BB * HH * 4);
    float* ctxbuf  = (float*)carve((size_t)BB * HH * 4);
    float* att_all = (float*)carve((size_t)TP1 * BB * HH * 4);

    // ---- output layout: outputs[B,51,V] | atts[51,S,B,1] | hF[B,H] | cF[B,H]
    float* out_logits = (float*)d_out;
    float* out_atts   = out_logits + (long)BB * TP1 * VV;
    float* out_hF     = out_atts + (long)TP1 * SS * BB;
    float* out_cF     = out_hF + (long)BB * HH;

    // ---- pack weights into WMMA B panels (bf16) ----
    launch_pack_b(stream, Wo, WoP, HH, VV);
    launch_pack_b(stream, Wl, WlP, EE + HH, 4 * HH);
    launch_pack_b(stream, W1, W1P, HH, HH);
    launch_pack_b(stream, W2, W2P, HH, HH);
    launch_pack_b(stream, W3, W3P, 2 * HH, HH);
    launch_pack_b(stream, W4, W4P, FF, HH);
    launch_pack_b(stream, W5, W5P, HH, HH);

    // ---- precomputed encoder/field projections (tanh) ----
    launch_pack_a(stream, enc, HH, enc, HH, HH, encAP, BB * SS, HH);
    launch_pack_a(stream, fld, FF, fld, FF, FF, fldAP, BB * SS, FF);
    launch_gemm(stream, encAP, W1P, b1, phi_h, HH, BB * SS, HH, HH, /*tanh*/1, 0);
    launch_gemm(stream, fldAP, W4P, b4, phi_f, HH, BB * SS, HH, FF, /*tanh*/1, 0);

    // ---- init recurrent state ----
    hipMemcpyAsync(hbuf, h0, (size_t)BB * HH * 4, hipMemcpyDeviceToDevice, stream);
    hipMemcpyAsync(cbuf, c0, (size_t)BB * HH * 4, hipMemcpyDeviceToDevice, stream);

    // ---- 51 sequential decode steps (vocab projection hoisted out) ----
    for (int t = 0; t < TP1; ++t) {
        const float* X;
        int xstride;
        if (t == 0) { X = emb + (long)2 * EE; xstride = 0; }            // <sos> broadcast
        else        { X = iemb + (long)(t - 1) * EE; xstride = TT * EE; }

        // z = [x ; h_prev] @ Wl + bl     [64,1024]@[1024,2048]
        launch_pack_a(stream, X, xstride, hbuf, HH, EE, zAP, BB, EE + HH);
        launch_gemm(stream, zAP, WlP, bl, zbuf, 4 * HH, BB, 4 * HH, EE + HH, 0, 0);
        hipLaunchKernelGGL(lstm_ew_kernel, dim3((BB * HH + 255) / 256), dim3(256), 0,
                           stream, zbuf, hbuf, cbuf);

        // pack h once; reused by both gamma (W2) and alpha (W5) projections
        launch_pack_a(stream, hbuf, HH, hbuf, HH, HH, hAP, BB, HH);
        launch_gemm(stream, hAP, W2P, b2, gbuf, HH, BB, HH, HH, 1, 0);
        launch_gemm(stream, hAP, W5P, b5, abuf, HH, BB, HH, HH, 1, 0);

        // dual softmax attention + context
        hipLaunchKernelGGL(attention_kernel, dim3(BB), dim3(256), 0, stream,
                           phi_h, phi_f, enc, gbuf, abuf, ctxbuf, out_atts, t);

        // att_out = [context ; h] @ W3 + b3 -> staged for the big vocab GEMM
        launch_pack_a(stream, ctxbuf, HH, hbuf, HH, HH, catAP, BB, 2 * HH);
        launch_gemm(stream, catAP, W3P, b3, att_all + (long)t * BB * HH, HH,
                    BB, HH, 2 * HH, 0, 0);
    }

    // ---- one large vocab projection: [3264,512]@[512,32000]+bo,
    // rows [t*64+b] remapped into outputs[b,t,:] ----
    launch_pack_a(stream, att_all, HH, att_all, HH, HH, attAP, TP1 * BB, HH);
    launch_gemm(stream, attAP, WoP, bo, out_logits, VV, TP1 * BB, VV, HH, 0, TP1);

    // ---- final state outputs ----
    hipMemcpyAsync(out_hF, hbuf, (size_t)BB * HH * 4, hipMemcpyDeviceToDevice, stream);
    hipMemcpyAsync(out_cF, cbuf, (size_t)BB * HH * 4, hipMemcpyDeviceToDevice, stream);
}
